// MultiBanditNet_60765197303882
// MI455X (gfx1250) — matrix operand order repaired and checked by reading, not compiled, not run on hardware
//
#include <hip/hip_runtime.h>

typedef __attribute__((ext_vector_type(16))) __bf16 v16bf;
typedef __attribute__((ext_vector_type(8)))  float  v8f;

__device__ __forceinline__ unsigned pack2bf(float a, float b) {
    unsigned ua = __float_as_uint(a), ub = __float_as_uint(b);
    ua = (ua + 0x7FFFu + ((ua >> 16) & 1u)) >> 16;   // RNE truncate to bf16
    ub = (ub + 0x7FFFu + ((ub >> 16) & 1u)) >> 16;
    return (ub << 16) | ua;
}
__device__ __forceinline__ float bf2f(unsigned short h) {
    return __uint_as_float(((unsigned)h) << 16);
}

// ---------------- per-expert row bucketing ----------------
__global__ void init_counts_kernel(int* cnt) {
    if (threadIdx.x < 8) cnt[threadIdx.x] = 0;
}

__global__ void scatter_kernel(const int* __restrict__ sel, int* __restrict__ cnt,
                               int* __restrict__ rows, int Bn) {
    int r = blockIdx.x * 256 + threadIdx.x;
    if (r < Bn) {
        int e = sel[r];
        int p = atomicAdd(&cnt[e], 1);
        rows[e * Bn + p] = r;
    }
}

// ---------------- bf16 WMMA GEMM: Out = act(A @ Bw + bias) ----------------
// Block tile 64x256, 8 waves (4x2), wave tile 16x128 (8 wmma accumulators),
// k-step 32, double-buffered LDS software pipeline.
// LDS layouts (packed bf16 pairs as dwords):
//   A: [row][kpair]  stride 20  -> per-lane fragment = 2x ds_load_b128
//   B: [col][kpair]  stride 20  -> per-lane fragment = 2x ds_load_b128 (conflict-free)
// bf16-A GEMMs stage A via global_load_async_to_lds_b32 (ASYNCcnt path).
template <bool ABF16, bool OUTBF16, bool GATHER>
__global__ __launch_bounds__(256) void gemm_wmma_kernel(
    const void* __restrict__ Ap, const float* __restrict__ Bw,
    const float* __restrict__ bias, void* __restrict__ Outp,
    int M, int N, int K, int lda, int ldb, int ldo,
    const int* __restrict__ rowlist, const int* __restrict__ counts, int Bcap) {
    __shared__ unsigned alds[2][64 * 20];    // 10 KB
    __shared__ unsigned blds[2][256 * 20];   // 40 KB

    const int tid = threadIdx.x;
    const int mt = blockIdx.x, nt = blockIdx.y;
    int Me = M;
    const int* rl = nullptr;
    if constexpr (GATHER) {
        const int ex = blockIdx.z;
        Me = counts[ex];
        if (mt * 64 >= Me) return;  // uniform across block: EXEC stays full
        rl = rowlist + (size_t)ex * Bcap;
        Bw += (size_t)ex * K * ldb;
        bias += (size_t)ex * N;
    }

    const float* aF = (const float*)Ap;
    const unsigned short* aH = (const unsigned short*)Ap;

    // Resolve the 4 A-rows this thread stages each k-step (fixed across loop).
    // Clamped to row 0 when past Me: garbage only feeds tail accumulator rows,
    // which are never stored (WMMA D rows depend only on matching A rows).
    int arow[4];
    bool aval[4];
#pragma unroll
    for (int i = 0; i < 4; i++) {
        int idx = tid + 256 * i;
        int gm = mt * 64 + (idx >> 4);
        aval[i] = (gm < Me);
        int gmc = aval[i] ? gm : 0;
        arow[i] = GATHER ? rl[gmc] : gmc;
    }

    const int bcol = nt * 256 + tid;          // this thread's B column
    const float* bp0 = Bw + bcol;

    float ast[8];    // staged A (f32 path)
    float bst[32];   // staged B

    // ---- staging phases ----
    auto loadA = [&](int k0) {
        if constexpr (!ABF16) {
#pragma unroll
            for (int i = 0; i < 4; i++) {
                int kp = (tid + 256 * i) & 15;
                const float* p = aF + (size_t)arow[i] * lda + (k0 + 2 * kp);
                ast[2 * i] = p[0];
                ast[2 * i + 1] = p[1];
            }
        }
    };
    auto asyncA = [&](int buf, int k0) {
        if constexpr (ABF16) {
#pragma unroll
            for (int i = 0; i < 4; i++) {
                int idx = tid + 256 * i;
                int r = idx >> 4, kp = idx & 15;
                const unsigned short* gp = aH + (size_t)arow[i] * lda + (k0 + 2 * kp);
                unsigned lb = (unsigned)(uintptr_t)&alds[buf][r * 20 + kp];
                asm volatile("global_load_async_to_lds_b32 %0, %1, off"
                             :: "v"(lb), "v"(gp) : "memory");
            }
        }
    };
    auto storeA = [&](int buf) {
        if constexpr (!ABF16) {
#pragma unroll
            for (int i = 0; i < 4; i++) {
                int idx = tid + 256 * i;
                int r = idx >> 4, kp = idx & 15;
                unsigned v = pack2bf(ast[2 * i], ast[2 * i + 1]);
                alds[buf][r * 20 + kp] = aval[i] ? v : 0u;
            }
        }
    };
    auto loadB = [&](int k0) {
#pragma unroll
        for (int kp = 0; kp < 16; kp++) {
            int kk = k0 + 2 * kp;
            bst[2 * kp] = bp0[(size_t)kk * ldb];
            bst[2 * kp + 1] = bp0[(size_t)(kk + 1) * ldb];
        }
    };
    auto storeB = [&](int buf) {
#pragma unroll
        for (int kp = 0; kp < 16; kp++)
            blds[buf][tid * 20 + kp] = pack2bf(bst[2 * kp], bst[2 * kp + 1]);
    };

    const int w = tid >> 5, lane = tid & 31;
    const int wm = w >> 1, wn = w & 1;
    const int l15 = lane & 15;
    const bool lo = lane < 16;

    const v8f vzero = {0.f, 0.f, 0.f, 0.f, 0.f, 0.f, 0.f, 0.f};
    v8f acc[8] = {vzero, vzero, vzero, vzero, vzero, vzero, vzero, vzero};

    // prologue: fill buffer 0
    loadA(0);
    asyncA(0, 0);
    loadB(0);
    storeA(0);
    storeB(0);
    if constexpr (ABF16) asm volatile("s_wait_asynccnt 0x0" ::: "memory");
    __syncthreads();

    const int nk = K >> 5;
    for (int ks = 0; ks < nk; ++ks) {
        const int cur = ks & 1;
        const bool more = (ks + 1) < nk;
        if (more) {  // issue next tile's global loads before compute
            loadA((ks + 1) << 5);
            asyncA(cur ^ 1, (ks + 1) << 5);
            loadB((ks + 1) << 5);
        }
        // ---- compute current buffer ----
        union UA { unsigned u[8]; v16bf v; } ua;
        {
            const unsigned* ap = &alds[cur][(wm * 16 + l15) * 20];
            const int kb0 = lo ? 0 : 8, kb1 = lo ? 4 : 12;
#pragma unroll
            for (int j = 0; j < 4; j++) {
                ua.u[j] = ap[kb0 + j];
                ua.u[4 + j] = ap[kb1 + j];
            }
        }
#pragma unroll
        for (int s = 0; s < 8; s++) {
            union UB { unsigned u[8]; v16bf v; } ub;
            const unsigned* bp = &blds[cur][(wn * 128 + s * 16 + l15) * 20 + (lo ? 0 : 8)];
#pragma unroll
            for (int j = 0; j < 8; j++) ub.u[j] = bp[j];
            acc[s] = __builtin_amdgcn_wmma_f32_16x16x32_bf16(
                false, ua.v, false, ub.v, (short)0, acc[s], false, false);
        }
        if (more) {  // convert + park into next buffer after compute issued
            storeA(cur ^ 1);
            storeB(cur ^ 1);
        }
        if constexpr (ABF16) asm volatile("s_wait_asynccnt 0x0" ::: "memory");
        __syncthreads();
    }

    // ---- epilogue: C/D layout -> lane<16: M=i, lane>=16: M=i+8; N=lane&15 ----
    int grows[8];
#pragma unroll
    for (int i = 0; i < 8; i++) {
        int lm = mt * 64 + wm * 16 + (lo ? i : i + 8);
        grows[i] = (lm < Me) ? (GATHER ? rl[lm] : lm) : -1;
    }
    unsigned short* oH = (unsigned short*)Outp;
    float* oF = (float*)Outp;
#pragma unroll
    for (int s = 0; s < 8; s++) {
        int nn = nt * 256 + wn * 128 + s * 16 + l15;
        float bv = bias[nn];
#pragma unroll
        for (int i = 0; i < 8; i++) {
            int grow = grows[i];
            if (grow < 0) continue;
            float v = acc[s][i] + bv;
            if constexpr (OUTBF16) {
                v = fmaxf(v, 0.f);  // relu
                unsigned uv = __float_as_uint(v);
                uv = (uv + 0x7FFFu + ((uv >> 16) & 1u)) >> 16;
                oH[(size_t)grow * ldo + nn] = (unsigned short)uv;
            } else {
                oF[(size_t)grow * ldo + nn] = v;
            }
        }
    }
}

// ---------------- small heads: option softmax (E=8) + termination sigmoid ----------------
__global__ __launch_bounds__(256) void heads_kernel(
    const unsigned short* __restrict__ Hs, const unsigned short* __restrict__ Ht,
    const float* __restrict__ Wo2, const float* __restrict__ bo2,
    const float* __restrict__ Wt2, const float* __restrict__ bt2,
    const int* __restrict__ sel, float* __restrict__ opt, float* __restrict__ term,
    float* __restrict__ selopt) {
    const int wv = threadIdx.x >> 5, lane = threadIdx.x & 31;
    const int row = blockIdx.x * 8 + wv;
    const unsigned short* hs = Hs + (size_t)row * 1024;
    const unsigned short* ht = Ht + (size_t)row * 1024;
    float acc[8] = {0.f, 0.f, 0.f, 0.f, 0.f, 0.f, 0.f, 0.f};
    float ta = 0.f;
    for (int k = lane; k < 1024; k += 32) {
        float h = bf2f(hs[k]);
        const float* wrow = Wo2 + (size_t)k * 8;
#pragma unroll
        for (int e = 0; e < 8; e++) acc[e] += h * wrow[e];
        ta += bf2f(ht[k]) * Wt2[k];
    }
#pragma unroll
    for (int e = 0; e < 8; e++)
        for (int o = 16; o > 0; o >>= 1) acc[e] += __shfl_xor(acc[e], o, 32);
    for (int o = 16; o > 0; o >>= 1) ta += __shfl_xor(ta, o, 32);
    if (lane == 0) {
        float l[8], mx = -3.0e38f;
#pragma unroll
        for (int e = 0; e < 8; e++) { l[e] = acc[e] + bo2[e]; mx = fmaxf(mx, l[e]); }
        float s = 0.f;
#pragma unroll
        for (int e = 0; e < 8; e++) { l[e] = expf(l[e] - mx); s += l[e]; }
        float inv = 1.f / s;
#pragma unroll
        for (int e = 0; e < 8; e++) opt[(size_t)row * 8 + e] = l[e] * inv;
        term[row] = 1.f / (1.f + expf(-(ta + bt2[0])));
        selopt[row] = (float)sel[row];
    }
}

// ---------------- action softmax over A=512 + first-index argmax ----------------
__global__ __launch_bounds__(256) void action_softmax_kernel(
    const float* __restrict__ logits, float* __restrict__ probs,
    float* __restrict__ actout) {
    const int row = blockIdx.x, t = threadIdx.x;
    const float* lg = logits + (size_t)row * 512;
    float v0 = lg[t], v1 = lg[t + 256];
    __shared__ float red[256];
    __shared__ int redi[256];
    float tm = fmaxf(v0, v1);
    red[t] = tm;
    __syncthreads();
    for (int s = 128; s > 0; s >>= 1) {
        if (t < s) red[t] = fmaxf(red[t], red[t + s]);
        __syncthreads();
    }
    float rmax = red[0];
    __syncthreads();
    float e0 = expf(v0 - rmax), e1 = expf(v1 - rmax);
    red[t] = e0 + e1;
    __syncthreads();
    for (int s = 128; s > 0; s >>= 1) {
        if (t < s) red[t] += red[t + s];
        __syncthreads();
    }
    float inv = 1.f / red[0];
    __syncthreads();
    probs[(size_t)row * 512 + t] = e0 * inv;
    probs[(size_t)row * 512 + t + 256] = e1 * inv;
    // argmax (first max index, matching jnp.argmax)
    int ti = (v1 > v0) ? (t + 256) : t;
    red[t] = tm;
    redi[t] = ti;
    __syncthreads();
    for (int s = 128; s > 0; s >>= 1) {
        if (t < s) {
            float a = red[t], b = red[t + s];
            if (b > a || (b == a && redi[t + s] < redi[t])) {
                red[t] = b;
                redi[t] = redi[t + s];
            }
        }
        __syncthreads();
    }
    if (t == 0) actout[row] = (float)redi[0];
}

extern "C" void kernel_launch(void* const* d_in, const int* in_sizes, int n_in,
                              void* d_out, int out_size, void* d_ws, size_t ws_size,
                              hipStream_t stream) {
    const int Bn = 4096, Dd = 1024, Ee = 8, Hh = 1024, Aa = 512;
    const float* state = (const float*)d_in[0];
    const int* sel = (const int*)d_in[1];
    const float* Wo1 = (const float*)d_in[2];
    const float* bo1 = (const float*)d_in[3];
    const float* Wo2 = (const float*)d_in[4];
    const float* bo2 = (const float*)d_in[5];
    const float* We1 = (const float*)d_in[6];
    const float* be1 = (const float*)d_in[7];
    const float* We2 = (const float*)d_in[8];
    const float* be2 = (const float*)d_in[9];
    const float* We3 = (const float*)d_in[10];
    const float* be3 = (const float*)d_in[11];
    const float* Wt1 = (const float*)d_in[12];
    const float* bt1 = (const float*)d_in[13];
    const float* Wt2 = (const float*)d_in[14];
    const float* bt2 = (const float*)d_in[15];

    float* out = (float*)d_out;
    float* opt_out = out;                                // (B,E)
    float* act_out = opt_out + (size_t)Bn * Ee;          // (B,A)
    float* term_out = act_out + (size_t)Bn * Aa;         // (B,1)
    float* selopt_out = term_out + Bn;                   // (B,)
    float* selact_out = selopt_out + Bn;                 // (B,)

    // workspace carve-out (~42 MB total)
    char* ws = (char*)d_ws;
    size_t off = 0;
    auto carve = [&](size_t bytes) {
        size_t o = off;
        off = (off + bytes + 255) & ~(size_t)255;
        return o;
    };
    int* cnt = (int*)(ws + carve((size_t)Ee * 4));
    int* rows = (int*)(ws + carve((size_t)Ee * Bn * 4));
    unsigned short* Hs = (unsigned short*)(ws + carve((size_t)Bn * Hh * 2));
    unsigned short* Ht = (unsigned short*)(ws + carve((size_t)Bn * Hh * 2));
    unsigned short* h1 = (unsigned short*)(ws + carve((size_t)Bn * Hh * 2));
    unsigned short* h2 = (unsigned short*)(ws + carve((size_t)Bn * Hh * 2));
    float* logits = (float*)(ws + carve((size_t)Bn * Aa * 4));
    (void)ws_size; (void)in_sizes; (void)n_in; (void)out_size;

    // 1) bucket rows by selected expert
    init_counts_kernel<<<1, 64, 0, stream>>>(cnt);
    scatter_kernel<<<Bn / 256, 256, 0, stream>>>(sel, cnt, rows, Bn);

    // 2) big dense GEMMs for option/termination hidden layers (relu -> bf16)
    gemm_wmma_kernel<false, true, false><<<dim3(Bn / 64, Hh / 256), 256, 0, stream>>>(
        state, Wo1, bo1, Hs, Bn, Hh, Dd, Dd, Hh, Hh, nullptr, nullptr, 0);
    gemm_wmma_kernel<false, true, false><<<dim3(Bn / 64, Hh / 256), 256, 0, stream>>>(
        state, Wt1, bt1, Ht, Bn, Hh, Dd, Dd, Hh, Hh, nullptr, nullptr, 0);

    // 3) tiny heads: option softmax (E=8) + termination sigmoid + copy selected_options
    heads_kernel<<<Bn / 8, 256, 0, stream>>>(Hs, Ht, Wo2, bo2, Wt2, bt2, sel,
                                             opt_out, term_out, selopt_out);

    // 4) expert path: only the selected expert per row (8x FLOP cut)
    gemm_wmma_kernel<false, true, true><<<dim3(Bn / 64, Hh / 256, Ee), 256, 0, stream>>>(
        state, We1, be1, h1, Bn, Hh, Dd, Dd, Hh, Hh, rows, cnt, Bn);
    gemm_wmma_kernel<true, true, true><<<dim3(Bn / 64, Hh / 256, Ee), 256, 0, stream>>>(
        h1, We2, be2, h2, Bn, Hh, Hh, Hh, Hh, Hh, rows, cnt, Bn);
    gemm_wmma_kernel<true, false, true><<<dim3(Bn / 64, Aa / 256, Ee), 256, 0, stream>>>(
        h2, We3, be3, logits, Bn, Aa, Hh, Hh, Aa, Aa, rows, cnt, Bn);

    // 5) action softmax + argmax
    action_softmax_kernel<<<Bn, 256, 0, stream>>>(logits, act_out, selact_out);
}
